// EigenLayer_21268678050323
// MI455X (gfx1250) — matrix operand order repaired and checked by reading, not compiled, not run on hardware
//
#include <hip/hip_runtime.h>
#include <hip/hip_bf16.h>
#include <math.h>

// Problem constants (from reference)
#define BATCH 2048
#define DIM   2048
#define EDIM  64
#define NOUT  (EDIM * EDIM)   // 4096
#define EPS   1e-8f
#define NSWEEP 12

#define XN ((size_t)BATCH * DIM)   // 4,194,304 elements of x
#define WN ((size_t)NOUT * DIM)    // 8,388,608 elements of W

typedef __attribute__((ext_vector_type(16))) __bf16 v16bf;
typedef __attribute__((ext_vector_type(8)))  __bf16 v8bf;
typedef __attribute__((ext_vector_type(4)))  __bf16 v4bf;
typedef __attribute__((ext_vector_type(8)))  float  v8f;

// ---------------------------------------------------------------------------
// Kernel 0: one-time fp32 -> (bf16 hi, bf16 lo) plane split.
//   src ~ f32[n], dsth/dstl ~ bf16[n].  n multiple of 4; float4 vectorized.
// Bandwidth-bound (~100 MB total traffic ≈ 4 us at 23.3 TB/s).
// ---------------------------------------------------------------------------
__global__ __launch_bounds__(256) void split_bf16_planes(
    const float* __restrict__ src, __bf16* __restrict__ dsth,
    __bf16* __restrict__ dstl, size_t n4)
{
    size_t stride = (size_t)gridDim.x * blockDim.x;
    for (size_t i = blockIdx.x * blockDim.x + threadIdx.x; i < n4; i += stride) {
        float4 f = ((const float4*)src)[i];
        v4bf h, l;
        __bf16 h0 = (__bf16)f.x; h[0] = h0; l[0] = (__bf16)(f.x - (float)h0);
        __bf16 h1 = (__bf16)f.y; h[1] = h1; l[1] = (__bf16)(f.y - (float)h1);
        __bf16 h2 = (__bf16)f.z; h[2] = h2; l[2] = (__bf16)(f.z - (float)h2);
        __bf16 h3 = (__bf16)f.w; h[3] = h3; l[3] = (__bf16)(f.w - (float)h3);
        ((v4bf*)dsth)[i] = h;
        ((v4bf*)dstl)[i] = l;
    }
}

// Load a 16-element bf16 WMMA fragment as two aligned 16-byte halves.
__device__ __forceinline__ v16bf load_frag(const __bf16* p) {
    v8bf lo = *(const v8bf*)(p);
    v8bf hi = *(const v8bf*)(p + 16);
    return __builtin_shufflevector(lo, hi, 0, 1, 2, 3, 4, 5, 6, 7,
                                   8, 9, 10, 11, 12, 13, 14, 15);
}

// ---------------------------------------------------------------------------
// Kernel 1: m = x @ W^T via bf16 hi/lo-split WMMA, operands pre-split.
// Each wave computes a 32(M) x 64(N) strip = 2x4 tiles of 16x16.
//   per K-step (32): 24 WMMAs, 24 x b128 loads/lane -> ~32 FLOP/B from L2.
// Jobs: 64 M-strips * 64 N-strips = 4096 waves = 512 blocks * 8 waves.
// ---------------------------------------------------------------------------
__global__ __launch_bounds__(256) void eig_gemm_wmma(
    const __bf16* __restrict__ xh, const __bf16* __restrict__ xl,
    const __bf16* __restrict__ wh, const __bf16* __restrict__ wl,
    float* __restrict__ m)
{
    const int lane  = threadIdx.x & 31;
    const int wave  = threadIdx.x >> 5;
    const int job    = blockIdx.x * 8 + wave;
    const int mStrip = job >> 6;          // 0..63 (rows of 32)
    const int nStrip = job & 63;          // 0..63 (cols of 64)

    const int hi    = lane >> 4;          // half-wave select
    const int l15   = lane & 15;
    const int aKoff = hi * 8;             // A fragment K base: 0 or 8
    const int bKoff = hi * 8;             // B fragment K base: 0 or 8

    // A rows for the two M-tiles; B rows (= N columns) for the four N-tiles.
    const size_t rowA0 = (size_t)(mStrip * 32 + l15)      * DIM;
    const size_t rowA1 = (size_t)(mStrip * 32 + 16 + l15) * DIM;
    const size_t rowB0 = (size_t)(nStrip * 64 + l15)      * DIM;
    const size_t rowB1 = rowB0 + (size_t)16 * DIM;
    const size_t rowB2 = rowB0 + (size_t)32 * DIM;
    const size_t rowB3 = rowB0 + (size_t)48 * DIM;

    v8f acc[2][4];
    #pragma unroll
    for (int i = 0; i < 2; ++i)
        #pragma unroll
        for (int j = 0; j < 4; ++j) acc[i][j] = (v8f){};

    for (int k0 = 0; k0 < DIM; k0 += 32) {
        // ---- A fragments (hi & lo planes), per 16-bit A 16x32 layout:
        //      lanes 0-15: K = {0..7, 16..23}; lanes 16-31: K = {8..15, 24..31}
        v16bf aH[2], aL[2];
        {
            const size_t o0 = rowA0 + k0 + aKoff;
            const size_t o1 = rowA1 + k0 + aKoff;
            // two b128 halves per fragment: K base and K base+16
            v8bf h0a = *(const v8bf*)(xh + o0), h0b = *(const v8bf*)(xh + o0 + 16);
            v8bf l0a = *(const v8bf*)(xl + o0), l0b = *(const v8bf*)(xl + o0 + 16);
            v8bf h1a = *(const v8bf*)(xh + o1), h1b = *(const v8bf*)(xh + o1 + 16);
            v8bf l1a = *(const v8bf*)(xl + o1), l1b = *(const v8bf*)(xl + o1 + 16);
            aH[0] = __builtin_shufflevector(h0a, h0b, 0,1,2,3,4,5,6,7,8,9,10,11,12,13,14,15);
            aL[0] = __builtin_shufflevector(l0a, l0b, 0,1,2,3,4,5,6,7,8,9,10,11,12,13,14,15);
            aH[1] = __builtin_shufflevector(h1a, h1b, 0,1,2,3,4,5,6,7,8,9,10,11,12,13,14,15);
            aL[1] = __builtin_shufflevector(l1a, l1b, 0,1,2,3,4,5,6,7,8,9,10,11,12,13,14,15);
        }
        // ---- B fragments, per 16-bit B 32x16 layout:
        //      lanes 0-15: K = {0..7, 16..23}; lanes 16-31: K = {8..15, 24..31}
        v16bf bH[4], bL[4];
        {
            const size_t o0 = rowB0 + k0 + bKoff;
            const size_t o1 = rowB1 + k0 + bKoff;
            const size_t o2 = rowB2 + k0 + bKoff;
            const size_t o3 = rowB3 + k0 + bKoff;
            bH[0] = load_frag(wh + o0);  bL[0] = load_frag(wl + o0);
            bH[1] = load_frag(wh + o1);  bL[1] = load_frag(wl + o1);
            bH[2] = load_frag(wh + o2);  bL[2] = load_frag(wl + o2);
            bH[3] = load_frag(wh + o3);  bL[3] = load_frag(wl + o3);
        }
        // ---- 3-term bf16 emulation of fp32 product, f32 accumulate ----
        #pragma unroll
        for (int i = 0; i < 2; ++i) {
            #pragma unroll
            for (int j = 0; j < 4; ++j) {
                acc[i][j] = __builtin_amdgcn_wmma_f32_16x16x32_bf16(
                    false, aH[i], false, bH[j], (short)0, acc[i][j], false, false);
                acc[i][j] = __builtin_amdgcn_wmma_f32_16x16x32_bf16(
                    false, aH[i], false, bL[j], (short)0, acc[i][j], false, false);
                acc[i][j] = __builtin_amdgcn_wmma_f32_16x16x32_bf16(
                    false, aL[i], false, bH[j], (short)0, acc[i][j], false, false);
            }
        }
    }

    // ---- store C: VGPR r -> M = r (lanes 0-15) / r+8 (lanes 16-31) ----
    #pragma unroll
    for (int i = 0; i < 2; ++i) {
        const int rbase = mStrip * 32 + i * 16 + hi * 8;
        #pragma unroll
        for (int j = 0; j < 4; ++j) {
            const int col = nStrip * 64 + j * 16 + l15;
            #pragma unroll
            for (int r = 0; r < 8; ++r)
                m[(size_t)(rbase + r) * NOUT + col] = acc[i][j][r];
        }
    }
}

// ---------------------------------------------------------------------------
// Kernel 2: per batch element (one 256-thread block per matrix):
//   A = 0.5*((m+bias) + (m+bias)^T) + reg*I   (fused into the LDS load)
//   two-sided cyclic Jacobi (round-robin pivots, 32 disjoint pairs/step,
//   63 steps/sweep, NSWEEP sweeps), eigenvalues = diagonal,
//   rank-sort descending, out = log(max(lambda, EPS)).
// ---------------------------------------------------------------------------
__global__ __launch_bounds__(256) void eig_jacobi(
    const float* __restrict__ m, const float* __restrict__ bias,
    const float* __restrict__ regp, float* __restrict__ out)
{
    __shared__ float A[EDIM][EDIM + 1];   // +1 pad: spread LDS banks
    __shared__ float cs[32], sn[32];
    __shared__ float vals[EDIM];

    const int t = threadIdx.x;
    const int b = blockIdx.x;
    const float reg = regp[0];
    const float* __restrict__ g = m + (size_t)b * NOUT;

    // Load + bias + symmetrize + regularize diagonal
    for (int e = t; e < NOUT; e += 256) {
        const int i = e >> 6, j = e & 63;
        float v = 0.5f * ((g[e] + bias[e]) + (g[j * EDIM + i] + bias[j * EDIM + i]));
        if (i == j) v += reg;
        A[i][j] = v;
    }
    __syncthreads();

    for (int sweep = 0; sweep < NSWEEP; ++sweep) {
        for (int r = 0; r < 63; ++r) {
            // --- compute 32 disjoint rotations (threads 0..31) ---
            if (t < 32) {
                const int k  = t;
                const int pp = (r + k) % 63;
                const int qq = (k == 0) ? 63 : (r + 63 - k) % 63;
                const int p = pp < qq ? pp : qq;
                const int q = pp < qq ? qq : pp;
                const float app = A[p][p], aqq = A[q][q], apq = A[p][q];
                float co = 1.0f, si = 0.0f;
                if (fabsf(apq) > 1e-36f) {
                    const float tau = (aqq - app) / (2.0f * apq);
                    const float tt  = copysignf(1.0f, tau) /
                                      (fabsf(tau) + sqrtf(1.0f + tau * tau));
                    co = 1.0f / sqrtf(1.0f + tt * tt);
                    si = tt * co;
                }
                cs[k] = co; sn[k] = si;
            }
            __syncthreads();

            // pair assignment for update phases: 8 threads per pair, 8 cols each
            const int k  = t >> 3;
            const int pp = (r + k) % 63;
            const int qq = (k == 0) ? 63 : (r + 63 - k) % 63;
            const int p = pp < qq ? pp : qq;
            const int q = pp < qq ? qq : pp;
            const float co = cs[k], si = sn[k];
            const int j0 = (t & 7) * 8;

            // --- row phase: (row_p,row_q) <- (c*rp - s*rq, s*rp + c*rq) ---
            #pragma unroll
            for (int j = 0; j < 8; ++j) {
                const int jc = j0 + j;
                const float ap = A[p][jc], aq = A[q][jc];
                A[p][jc] = co * ap - si * aq;
                A[q][jc] = si * ap + co * aq;
            }
            __syncthreads();

            // --- column phase ---
            #pragma unroll
            for (int j = 0; j < 8; ++j) {
                const int jc = j0 + j;
                const float ap = A[jc][p], aq = A[jc][q];
                A[jc][p] = co * ap - si * aq;
                A[jc][q] = si * ap + co * aq;
            }
            __syncthreads();
        }
    }

    // Eigenvalues = diagonal; sort descending by rank-count; log-clamp.
    if (t < EDIM) vals[t] = A[t][t];
    __syncthreads();
    if (t < EDIM) {
        const float v = vals[t];
        int rank = 0;
        #pragma unroll 8
        for (int j = 0; j < EDIM; ++j) {
            const float vj = vals[j];
            rank += (vj > v) || (vj == v && j < t);
        }
        out[(size_t)b * EDIM + rank] = logf(fmaxf(v, EPS));
    }
}

// ---------------------------------------------------------------------------
// Host launcher
//   d_ws layout (bytes):
//     [0,  32MB)  m scratch   f32[2048*4096]
//     [32, 40MB)  xh bf16[XN]
//     [40, 48MB)  xl bf16[XN]
//     [48, 64MB)  wh bf16[WN]
//     [64, 80MB)  wl bf16[WN]
// ---------------------------------------------------------------------------
extern "C" void kernel_launch(void* const* d_in, const int* in_sizes, int n_in,
                              void* d_out, int out_size, void* d_ws, size_t ws_size,
                              hipStream_t stream)
{
    const float* x    = (const float*)d_in[0];   // [2048, 2048]
    const float* W    = (const float*)d_in[1];   // [4096, 2048]
    const float* bias = (const float*)d_in[2];   // [4096]
    const float* reg  = (const float*)d_in[3];   // [1]
    float* out = (float*)d_out;                  // [2048, 64]

    char* ws = (char*)d_ws;
    float*  mtmp = (float*)ws;                                   // 32 MB
    __bf16* xh = (__bf16*)(ws + (size_t)32 * 1024 * 1024);       //  8 MB
    __bf16* xl = xh + XN;                                        //  8 MB
    __bf16* wh = xl + XN;                                        // 16 MB
    __bf16* wl = wh + WN;                                        // 16 MB

    // one-time hi/lo plane split (bandwidth-bound)
    split_bf16_planes<<<dim3(1024), dim3(256), 0, stream>>>(x, xh, xl, XN / 4);
    split_bf16_planes<<<dim3(2048), dim3(256), 0, stream>>>(W, wh, wl, WN / 4);

    // 4096 wave-jobs (64 M-strips x 64 N-strips), 8 waves per 256-thread block
    eig_gemm_wmma<<<dim3(512), dim3(256), 0, stream>>>(xh, xl, wh, wl, mtmp);

    // one block per 64x64 matrix
    eig_jacobi<<<dim3(BATCH), dim3(256), 0, stream>>>(mtmp, bias, reg, out);
}
